// LSTMClassifier_111669149898
// MI455X (gfx1250) — compile-verified, matrix operand order checked by reading, not run on hardware
//
#include <hip/hip_runtime.h>

// ---------------------------------------------------------------------------
// LSTM text classifier for MI455X (gfx1250, wave32, WMMA, async-LDS).
//   K1: f32 -> bf16 weight conversion
//   K2: embedding gather + bf16 convert            (bandwidth bound)
//   K3: xg = xe @ w_ih^T + bias  -> [T][B][4H]     (68.7 GFLOP, v_wmma bf16,
//                                                   async B-tile -> LDS)
//   K4: persistent single-WGP LSTM scan            (2048 sequential WMMA steps,
//                                                   double-buffered async xg)
//   K5: mean-pool -> ReLU MLP head
// ---------------------------------------------------------------------------

typedef __attribute__((ext_vector_type(16))) __bf16 bfvec16;
typedef __attribute__((ext_vector_type(8)))  __bf16 bfvec8;
typedef __attribute__((ext_vector_type(4)))  __bf16 bfvec4;
typedef __attribute__((ext_vector_type(8)))  float  fvec8;

#define B_    32
#define T_    2048
#define E_    512
#define H_    256
#define G_    1024   /* 4*H */
#define NCLS_ 20

// One wave-wide async 16B global->LDS copy (ASYNCcnt-tracked).
// NOTE: the LDS destination pointer is passed as an asm operand even though
// the instruction encodes only the 32-bit LDS offset — this forces the
// address of the LDS object to escape, so alias analysis keeps the later
// LDS reads (otherwise a static __shared__ array written only by this asm
// is "never written" and its loads fold to undef).
__device__ __forceinline__ void async_g2l_b128(unsigned lds_off,
                                               const void* gptr,
                                               void* lds_ptr) {
    unsigned long long ga = (unsigned long long)gptr;
    asm volatile("global_load_async_to_lds_b128 %0, %1, off"
                 :: "v"(lds_off), "v"(ga), "v"(lds_ptr) : "memory");
}
__device__ __forceinline__ void wait_async0() {
    asm volatile("s_wait_asynccnt 0x0" ::: "memory");
}

// ---------------- K1: weight conversion ----------------
__global__ void lstm_cvt_weights(const float* __restrict__ w_ih,
                                 const float* __restrict__ w_hh,
                                 __bf16* __restrict__ w_ih_b,
                                 __bf16* __restrict__ w_hh_b) {
    int i = blockIdx.x * 256 + threadIdx.x;
    const int n1 = G_ * E_;
    if (i < n1) w_ih_b[i] = (__bf16)w_ih[i];
    int j = i - n1;
    if (j >= 0 && j < G_ * H_) w_hh_b[j] = (__bf16)w_hh[j];
}

// ---------------- K2: embedding gather -> bf16 ----------------
__global__ void lstm_embed_gather(const int* __restrict__ x,
                                  const float* __restrict__ emb,
                                  __bf16* __restrict__ xe) {
    int row = blockIdx.x;                   // b*T + t
    int tok = x[row];
    const float4* src = (const float4*)(emb + (size_t)tok * E_);
    bfvec4* dst = (bfvec4*)(xe + (size_t)row * E_);
    float4 v = src[threadIdx.x];
    bfvec4 o;
    o.x = (__bf16)v.x; o.y = (__bf16)v.y; o.z = (__bf16)v.z; o.w = (__bf16)v.w;
    dst[threadIdx.x] = o;
}

// ---------------- K3: xg = xe @ w_ih^T + bias (bf16 WMMA) ----------------
// Block: 256 thr = 8 waves. Block tile: 128(M) x 64(N). Wave tile: 16 x 64.
// w_ih tile (64 rows x 512 K, bf16 = 64KB) streamed into LDS asynchronously.
// Output layout: xg[t][b][n]  (time-major for the scan kernel).
__global__ void __launch_bounds__(256)
lstm_gemm_xg(const __bf16* __restrict__ xe,
             const __bf16* __restrict__ wih,
             const float* __restrict__ b_ih,
             const float* __restrict__ b_hh,
             __bf16* __restrict__ xg) {
    __shared__ __bf16 Bs[64 * E_];          // 64 KB, LDS offset 0

    const int nblock = blockIdx.y * 64;
    // async contiguous 64KB copy (rows nblock..nblock+63 of w_ih) -> LDS
    {
        const char* gsrc = (const char*)(wih + (size_t)nblock * E_);
        char* lbase = (char*)Bs;
        #pragma unroll
        for (int i = 0; i < 16; ++i) {
            unsigned byt = (unsigned)((threadIdx.x + i * 256) * 16);
            async_g2l_b128(byt, gsrc + byt, lbase + byt);
        }
        wait_async0();
    }
    __syncthreads();

    const int wave = threadIdx.x >> 5;
    const int lane = threadIdx.x & 31;
    const int lr = lane & 15;               // row within 16-row fragment
    const int lh = lane >> 4;               // K-half select (contiguous k = 16*lh + s, same for A and B)
    const int m_base = blockIdx.x * 128 + wave * 16;

    fvec8 acc[4];
    #pragma unroll
    for (int ns = 0; ns < 4; ++ns)
        #pragma unroll
        for (int j = 0; j < 8; ++j) acc[ns][j] = 0.0f;

    const __bf16* arow = xe + (size_t)(m_base + lr) * E_ + lh * 16;
    for (int k = 0; k < E_; k += 32) {
        bfvec16 afrag = *(const bfvec16*)(arow + k);
        #pragma unroll
        for (int ns = 0; ns < 4; ++ns) {
            const __bf16* bptr = Bs + (ns * 16 + lr) * E_ + k + lh * 16;
            bfvec16 bfrag = *(const bfvec16*)bptr;
            acc[ns] = __builtin_amdgcn_wmma_f32_16x16x32_bf16(
                false, afrag, false, bfrag, (short)0, acc[ns], false, false);
        }
    }

    // epilogue: + (b_ih+b_hh), bf16, scatter to time-major xg[t][b][n]
    #pragma unroll
    for (int ns = 0; ns < 4; ++ns) {
        int n = nblock + ns * 16 + lr;
        float bias = b_ih[n] + b_hh[n];
        #pragma unroll
        for (int j = 0; j < 8; ++j) {
            int m = m_base + j + lh * 8;    // C rows: VGPR j -> M=j / M=8+j
            int b = m / T_, t = m % T_;     // row (b,t) of xe
            xg[((size_t)t * B_ + b) * G_ + n] = (__bf16)(acc[ns][j] + bias);
        }
    }
}

// ---------------- K4: persistent LSTM scan (single workgroup) ----------------
// 1024 thr = 32 waves. Per step: hW(32x1024) = h @ w_hh^T via 1024 WMMA
// 16x16x32 tiles; xg slab for step t+1 is async-prefetched into LDS
// (double-buffered) while step t computes; activations + c/h/pool in regs.
// Dynamic LDS: gates 64KB @0 | h 16KB @64K | xg[2] 128KB @80K  = 208KB.
#define LDSOFF_GATES 0u
#define LDSOFF_HBUF  65536u
#define LDSOFF_XG    81920u
__global__ void __launch_bounds__(1024)
lstm_scan(const __bf16* __restrict__ xg,   // [T][B][G] bf16
          const __bf16* __restrict__ whh,  // [G][H] bf16
          const int* __restrict__ x,
          float* __restrict__ pooled) {
    extern __shared__ char smem[];
    __bf16* gates = (__bf16*)(smem + LDSOFF_GATES);  // [32][1024]
    __bf16* hbuf  = (__bf16*)(smem + LDSOFF_HBUF);   // [32][256]
    __bf16* xgb   = (__bf16*)(smem + LDSOFF_XG);     // [2][32][1024]

    const int tid  = threadIdx.x;
    const int wave = tid >> 5;
    const int lane = tid & 31;
    const int lr = lane & 15;
    const int lh = lane >> 4;
    const int m_tile = wave >> 4;           // 0/1 -> batch rows 0-15 / 16-31
    const int n_base = (wave & 15) * 64;    // 64 gate columns per wave

    // epilogue ownership: thread -> (batch eb, 8 contiguous units jblk..+7)
    const int eb   = tid >> 5;
    const int jblk = (tid & 31) * 8;
    float c[8], pool[8];
    #pragma unroll
    for (int q = 0; q < 8; ++q) { c[q] = 0.0f; pool[q] = 0.0f; }
    float lenf = 0.0f;

    for (int i = tid; i < B_ * H_; i += 1024) hbuf[i] = (__bf16)0.0f;

    // prefetch slab t=0: 64KB = 1024 thr x 4 x 16B async copies
    {
        const char* g0 = (const char*)xg;
        #pragma unroll
        for (int i = 0; i < 4; ++i) {
            unsigned byt = (unsigned)(tid * 16 + i * 16384);
            async_g2l_b128(LDSOFF_XG + byt, g0 + byt, smem + LDSOFF_XG + byt);
        }
    }
    __syncthreads();

    for (int t = 0; t < T_; ++t) {
        wait_async0();                       // slab t resident in buffer t&1
        if (t + 1 < T_) {                    // prefetch t+1 under the compute
            const char* gn = (const char*)(xg + (size_t)(t + 1) * (B_ * G_));
            unsigned lbase = LDSOFF_XG + (((t + 1) & 1) ? 65536u : 0u);
            #pragma unroll
            for (int i = 0; i < 4; ++i) {
                unsigned byt = (unsigned)(tid * 16 + i * 16384);
                async_g2l_b128(lbase + byt, gn + byt, smem + lbase + byt);
            }
        }

        // --- hW = h(t-1) @ w_hh^T : 8 K-steps x 4 N-tiles of WMMA ---
        fvec8 acc[4];
        #pragma unroll
        for (int ns = 0; ns < 4; ++ns)
            #pragma unroll
            for (int j = 0; j < 8; ++j) acc[ns][j] = 0.0f;
        #pragma unroll
        for (int k = 0; k < H_; k += 32) {
            bfvec16 afrag =
                *(const bfvec16*)(hbuf + (m_tile * 16 + lr) * H_ + k + lh * 16);
            #pragma unroll
            for (int ns = 0; ns < 4; ++ns) {
                const __bf16* bptr =
                    whh + (size_t)(n_base + ns * 16 + lr) * H_ + k + lh * 16;
                bfvec16 bfrag = *(const bfvec16*)bptr;   // L2-resident stream
                acc[ns] = __builtin_amdgcn_wmma_f32_16x16x32_bf16(
                    false, afrag, false, bfrag, (short)0, acc[ns], false, false);
            }
        }
        // publish hW to LDS
        #pragma unroll
        for (int ns = 0; ns < 4; ++ns) {
            int n = n_base + ns * 16 + lr;
            #pragma unroll
            for (int j = 0; j < 8; ++j) {
                int b = m_tile * 16 + j + lh * 8;
                gates[b * G_ + n] = (__bf16)acc[ns][j];
            }
        }
        __syncthreads();   // gates visible + everyone's async wait passed

        // --- activations + state update (gate order i,f,g,o), vector LDS ---
        int tok = x[eb * T_ + t];
        float mask = (tok != 0) ? 1.0f : 0.0f;
        lenf += mask;
        const __bf16* gb = gates + eb * G_;
        const __bf16* xb = xgb + (t & 1) * (B_ * G_) + eb * G_;
        bfvec8 vi = *(const bfvec8*)(gb + jblk);
        bfvec8 vf = *(const bfvec8*)(gb + H_ + jblk);
        bfvec8 vg = *(const bfvec8*)(gb + 2 * H_ + jblk);
        bfvec8 vo = *(const bfvec8*)(gb + 3 * H_ + jblk);
        bfvec8 xi = *(const bfvec8*)(xb + jblk);
        bfvec8 xf = *(const bfvec8*)(xb + H_ + jblk);
        bfvec8 xgv = *(const bfvec8*)(xb + 2 * H_ + jblk);
        bfvec8 xo = *(const bfvec8*)(xb + 3 * H_ + jblk);
        bfvec8 hv;
        #pragma unroll
        for (int q = 0; q < 8; ++q) {
            float ig = (float)vi[q] + (float)xi[q];
            float fg = (float)vf[q] + (float)xf[q];
            float gg = (float)vg[q] + (float)xgv[q];
            float og = (float)vo[q] + (float)xo[q];
            ig = 1.0f / (1.0f + __expf(-ig));
            fg = 1.0f / (1.0f + __expf(-fg));
            gg = tanhf(gg);
            og = 1.0f / (1.0f + __expf(-og));
            c[q] = fg * c[q] + ig * gg;
            float h = og * tanhf(c[q]);
            pool[q] += mask * h;
            hv[q] = (__bf16)h;
        }
        *(bfvec8*)(hbuf + eb * H_ + jblk) = hv;   // single ds_store_b128
        __syncthreads();
    }

    float inv = 1.0f / fmaxf(lenf, 1.0f);
    #pragma unroll
    for (int q = 0; q < 8; ++q)
        pooled[eb * H_ + jblk + q] = pool[q] * inv;
}

// ---------------- K5: MLP head ----------------
__global__ void __launch_bounds__(512)
lstm_mlp_head(const float* __restrict__ pooled,
              const float* __restrict__ w1, const float* __restrict__ b1,
              const float* __restrict__ w2, const float* __restrict__ b2,
              float* __restrict__ out) {
    __shared__ float pr[H_];
    __shared__ float l1[2 * H_];
    const int b = blockIdx.x;
    for (int i = threadIdx.x; i < H_; i += 512) pr[i] = pooled[b * H_ + i];
    __syncthreads();

    const int j = threadIdx.x;
    float s = b1[j];
    const float* wr = w1 + (size_t)j * H_;
    #pragma unroll 4
    for (int k = 0; k < H_; ++k) s += pr[k] * wr[k];
    l1[j] = fmaxf(s, 0.0f);
    __syncthreads();

    if (j < NCLS_) {
        float s2 = b2[j];
        const float* w2r = w2 + (size_t)j * (2 * H_);
        #pragma unroll 4
        for (int k = 0; k < 2 * H_; ++k) s2 += l1[k] * w2r[k];
        out[b * NCLS_ + j] = s2;
    }
}

// ---------------- launcher ----------------
extern "C" void kernel_launch(void* const* d_in, const int* in_sizes, int n_in,
                              void* d_out, int out_size, void* d_ws, size_t ws_size,
                              hipStream_t stream) {
    (void)in_sizes; (void)n_in; (void)out_size; (void)ws_size;
    const int*   x    = (const int*)d_in[0];
    const float* emb  = (const float*)d_in[1];
    const float* w_ih = (const float*)d_in[2];
    const float* w_hh = (const float*)d_in[3];
    const float* b_ih = (const float*)d_in[4];
    const float* b_hh = (const float*)d_in[5];
    const float* w1   = (const float*)d_in[6];
    const float* b1   = (const float*)d_in[7];
    const float* w2   = (const float*)d_in[8];
    const float* b2   = (const float*)d_in[9];
    float* out = (float*)d_out;

    // workspace layout (bytes)
    char* ws = (char*)d_ws;
    __bf16* w_ih_b = (__bf16*)(ws + 0x0000000ull);   // 1 MB
    __bf16* w_hh_b = (__bf16*)(ws + 0x0100000ull);   // 0.5 MB
    __bf16* xe     = (__bf16*)(ws + 0x0200000ull);   // 64 MB  (B*T x E bf16)
    __bf16* xg     = (__bf16*)(ws + 0x4200000ull);   // 128 MB (T x B x 4H bf16)
    float*  pooled = (float*) (ws + 0xC200000ull);   // 32 KB  (B x H f32)

    hipLaunchKernelGGL(lstm_cvt_weights, dim3((G_ * E_ + G_ * H_) / 256),
                       dim3(256), 0, stream, w_ih, w_hh, w_ih_b, w_hh_b);

    hipLaunchKernelGGL(lstm_embed_gather, dim3(B_ * T_), dim3(128), 0, stream,
                       x, emb, xe);

    hipLaunchKernelGGL(lstm_gemm_xg, dim3((B_ * T_) / 128, G_ / 64),
                       dim3(256), 0, stream, xe, w_ih_b, b_ih, b_hh, xg);

    hipLaunchKernelGGL(lstm_scan, dim3(1), dim3(1024),
                       /*dyn LDS: gates 64K + h 16K + xg 2x64K*/ 212992,
                       stream, xg, w_hh_b, x, pooled);

    hipLaunchKernelGGL(lstm_mlp_head, dim3(B_), dim3(512), 0, stream,
                       pooled, w1, b1, w2, b2, out);
}
